// LSTM_60387240181862
// MI455X (gfx1250) — compile-verified, matrix operand order checked by reading, not run on hardware
//
#include <hip/hip_runtime.h>

// LSTM: B=256, T=512, IN=64, H=512, OUT=1, 2 layers.
#define Bn   256
#define Tn   512
#define INn  64
#define Hn   512
#define BH   (Bn * Hn)     // 131072 elements per time-slice of h
#define NWG  64            // persistent workgroups (4 batch tiles x 16 col tiles)
#define NSYNC 16           // workgroups per barrier domain (one batch tile)
#define NTHR 256           // 8 wave32 per workgroup

typedef __bf16 bf16_t;
typedef bf16_t bf16x8   __attribute__((ext_vector_type(8)));
typedef bf16_t bf16x16  __attribute__((ext_vector_type(16)));
typedef float  f32x8    __attribute__((ext_vector_type(8)));

__device__ __forceinline__ float sigm(float x) { return 1.0f / (1.0f + __expf(-x)); }

__device__ __forceinline__ bf16x16 join8(bf16x8 lo, bf16x8 hi) {
  return __builtin_shufflevector(lo, hi, 0, 1, 2, 3, 4, 5, 6, 7,
                                         8, 9, 10, 11, 12, 13, 14, 15);
}

// Monotonic barrier across the NSYNC workgroups of one batch tile.
// Counter zeroed by the init kernel at the start of every launch.
__device__ __forceinline__ void group_barrier(unsigned* cnt, unsigned& bar) {
  __syncthreads();
  ++bar;
  if (threadIdx.x == 0) {
    __threadfence();
    atomicAdd(cnt, 1u);
    const unsigned target = bar * NSYNC;
    volatile unsigned* vc = (volatile unsigned*)cnt;
    while (*vc < target) { __builtin_amdgcn_s_sleep(1); }
  }
  __syncthreads();
  __threadfence();
}

// One LSTM layer over all T steps. mode 0: h history == y0 buffer (layer 0
// writes its whole output sequence, reads t-1 slice as recurrent input).
// mode 1: ping-pong h buffers, store final h (f32) for the head GEMV.
__device__ __forceinline__ void layer_pass(
    bf16_t* __restrict__ wlds, int Kx,
    const float* __restrict__ Wih, const float* __restrict__ Whh,
    const float* __restrict__ bih, const float* __restrict__ bhh,
    const bf16_t* __restrict__ xbase, size_t xRowStride, size_t xTStride,
    bf16_t* __restrict__ y0, bf16_t* __restrict__ hbuf0, bf16_t* __restrict__ hbuf1,
    int mode, float* __restrict__ lastH, unsigned* __restrict__ cnt, unsigned& bar) {

  const int KP = Kx + Hn + 8;                 // padded LDS row stride (bank spread)
  const int batchBase = (blockIdx.x & 3) << 6;   // 4 batch tiles of 64
  const int colBase   = (blockIdx.x >> 2) << 5;  // 16 hidden-col tiles of 32
  const int wave  = threadIdx.x >> 5;
  const int lane  = threadIdx.x & 31;
  const int mSub  = (wave & 3) << 4;          // 16-row sub-tile per wave
  const int nSub  = (wave >> 2) << 4;         // 16-col sub-tile per wave
  const int laneN = lane & 15;
  const int kHi   = (lane >> 4) << 3;         // ISA 16-bit A/B half-wave K offset
  const int aRow  = batchBase + mSub + laneN; // A-fragment row (batch)
  const int jcol  = colBase + nSub + laneN;   // C column (hidden unit)

  // ---- Stage this tile's weight rows (4 gates x 32 cols, [Wih | Whh]) in LDS (bf16).
  {
    const int Ktot  = Kx + Hn;
    const int elems = 128 * Ktot;
    for (int idx = threadIdx.x; idx < elems; idx += NTHR) {
      int r = idx / Ktot;
      int k = idx - r * Ktot;
      int grow = ((r >> 5) << 9) + colBase + (r & 31);  // gate*512 + col
      float w = (k < Kx) ? Wih[(size_t)grow * Kx + k]
                         : Whh[(size_t)grow * Hn + (k - Kx)];
      wlds[(size_t)r * KP + k] = (bf16_t)w;
    }
  }
  __syncthreads();

  float biasv[4];
#pragma unroll
  for (int g = 0; g < 4; ++g)
    biasv[g] = bih[(g << 9) + jcol] + bhh[(g << 9) + jcol];

  f32x8 cstate;
#pragma unroll
  for (int r = 0; r < 8; ++r) cstate[r] = 0.0f;

  const int ktilesFull = (Kx + Hn) >> 5;
  const int ktilesX    = Kx >> 5;

  for (int t = 0; t < Tn; ++t) {
    const int tp = (t == 0) ? 0 : (t - 1);
    const bf16_t* xA = xbase + (size_t)aRow * xRowStride + (size_t)t * xTStride;
    const bf16_t* hrow;
    bf16_t* hstore;
    if (mode == 0) {
      hrow   = y0 + (size_t)tp * BH + (size_t)aRow * Hn;
      hstore = y0 + (size_t)t * BH;
    } else {
      hrow   = ((t & 1) ? hbuf1 : hbuf0) + (size_t)aRow * Hn;
      hstore = (t & 1) ? hbuf0 : hbuf1;
    }

    f32x8 acc[4];
#pragma unroll
    for (int g = 0; g < 4; ++g)
#pragma unroll
      for (int r = 0; r < 8; ++r) acc[g][r] = biasv[g];

    const int ktiles = (t == 0) ? ktilesX : ktilesFull;  // h0 == 0: skip recurrent K
    for (int kt = 0; kt < ktiles; ++kt) {
      const int k32 = kt << 5;
      const bf16_t* asrc = (k32 < Kx) ? (xA + k32) : (hrow + (k32 - Kx));
      bf16x8 a0 = *(const bf16x8*)(asrc + kHi);
      bf16x8 a1 = *(const bf16x8*)(asrc + 16 + kHi);
      bf16x16 af = join8(a0, a1);
#pragma unroll
      for (int g = 0; g < 4; ++g) {
        const bf16_t* wr = wlds + (size_t)((g << 5) + nSub + laneN) * KP + k32;
        bf16x8 b0 = *(const bf16x8*)(wr + kHi);
        bf16x8 b1 = *(const bf16x8*)(wr + 16 + kHi);
        bf16x16 bm = join8(b0, b1);
        acc[g] = __builtin_amdgcn_wmma_f32_16x16x32_bf16(
            false, af, false, bm, (short)0, acc[g], false, false);
      }
    }

    // LSTM cell elementwise; c stays in registers across all steps.
#pragma unroll
    for (int r = 0; r < 8; ++r) {
      float iv = sigm(acc[0][r]);
      float fv = sigm(acc[1][r]);
      float gv = tanhf(acc[2][r]);
      float ov = sigm(acc[3][r]);
      float cv = fv * cstate[r] + iv * gv;
      cstate[r] = cv;
      float hv = ov * tanhf(cv);
      int row = batchBase + mSub + kHi + r;   // C layout: lanes 16-31 hold M = r+8
      hstore[(size_t)row * Hn + jcol] = (bf16_t)hv;
      if (mode == 1 && t == Tn - 1) lastH[(size_t)row * Hn + jcol] = hv;
    }

    // Pull next step's x operand toward WGP$/L2 while we sit in the barrier.
    if (t + 1 < Tn) {
      __builtin_prefetch((const void*)(xA + xTStride), 0, 1);
      __builtin_prefetch((const void*)(xA + xTStride + 32), 0, 1);
    }
    group_barrier(cnt, bar);
  }
}

__global__ __launch_bounds__(NTHR, 1) void lstm_main(
    const float* __restrict__ Wih0, const float* __restrict__ Whh0,
    const float* __restrict__ bih0, const float* __restrict__ bhh0,
    const float* __restrict__ Wih1, const float* __restrict__ Whh1,
    const float* __restrict__ bih1, const float* __restrict__ bhh1,
    const bf16_t* __restrict__ xbf, bf16_t* __restrict__ y0,
    bf16_t* __restrict__ h1a, bf16_t* __restrict__ h1b,
    float* __restrict__ lastH, unsigned* __restrict__ cntBase) {
  // 128 weight rows x (512 + 512 + 8 pad) bf16 = 258 KB (CDNA5: 320 KB/WGP)
  __shared__ __align__(16) bf16_t wlds[128 * (Hn + Hn + 8)];
  // Per-batch-tile barrier counter (batch tiles are fully independent).
  unsigned* cnt = cntBase + (size_t)(blockIdx.x & 3) * 64;  // 256 B apart
  unsigned bar = 0;
  // Layer 0: x from bf16 copy of X, h history lives in y0 itself.
  layer_pass(wlds, INn, Wih0, Whh0, bih0, bhh0,
             xbf, (size_t)(Tn * INn), (size_t)INn,
             y0, nullptr, nullptr, 0, nullptr, cnt, bar);
  // Layer 1: x(t) = y0[t], recurrent h in ping-pong buffers.
  layer_pass(wlds, Hn, Wih1, Whh1, bih1, bhh1,
             y0, (size_t)Hn, (size_t)BH,
             y0, h1a, h1b, 1, lastH, cnt, bar);
}

__global__ void lstm_init(const float* __restrict__ X, bf16_t* __restrict__ xbf,
                          unsigned* __restrict__ cnt, long n) {
  long i = (long)blockIdx.x * blockDim.x + threadIdx.x;
  if (i < 4) cnt[i * 64] = 0u;          // zero the 4 barrier counters
  const long stride = (long)gridDim.x * blockDim.x;
  for (; i < n; i += stride) xbf[i] = (bf16_t)X[i];
}

__global__ void lstm_head(const float* __restrict__ lastH,
                          const float* __restrict__ Wl,
                          const float* __restrict__ bl,
                          float* __restrict__ out) {
  int b = threadIdx.x;  // 256 batches, OUT = 1
  float s = 0.0f;
  for (int j = 0; j < Hn; ++j) s += lastH[(size_t)b * Hn + j] * Wl[j];
  out[b] = s + bl[0];
}

extern "C" void kernel_launch(void* const* d_in, const int* in_sizes, int n_in,
                              void* d_out, int out_size, void* d_ws, size_t ws_size,
                              hipStream_t stream) {
  (void)in_sizes; (void)n_in; (void)out_size; (void)ws_size;
  const float* X    = (const float*)d_in[0];
  const float* Wih0 = (const float*)d_in[1];
  const float* Whh0 = (const float*)d_in[2];
  const float* bih0 = (const float*)d_in[3];
  const float* bhh0 = (const float*)d_in[4];
  const float* Wih1 = (const float*)d_in[5];
  const float* Whh1 = (const float*)d_in[6];
  const float* bih1 = (const float*)d_in[7];
  const float* bhh1 = (const float*)d_in[8];
  const float* Wl   = (const float*)d_in[9];
  const float* bl   = (const float*)d_in[10];

  char* ws = (char*)d_ws;
  // ws layout (all offsets 256B aligned)
  unsigned* cnt  = (unsigned*)ws;                                    // 4 counters, 256 B apart
  bf16_t*  xbf   = (bf16_t*)(ws + 1024);                             // 16 MiB  (B*T*IN bf16)
  bf16_t*  y0    = (bf16_t*)(ws + 1024 + 16777216);                  // 128 MiB (T*B*H bf16)
  bf16_t*  h1a   = (bf16_t*)(ws + 1024 + 16777216 + 134217728);      // 256 KiB
  bf16_t*  h1b   = (bf16_t*)(ws + 1024 + 16777216 + 134217728 + 262144);
  float*   lastH = (float*) (ws + 1024 + 16777216 + 134217728 + 524288);
  float*   out   = (float*)d_out;

  lstm_init<<<1024, 256, 0, stream>>>(X, xbf, cnt, (long)Bn * Tn * INn);
  lstm_main<<<NWG, NTHR, 0, stream>>>(Wih0, Whh0, bih0, bhh0,
                                      Wih1, Whh1, bih1, bhh1,
                                      xbf, y0, h1a, h1b, lastH, cnt);
  lstm_head<<<1, 256, 0, stream>>>(lastH, Wl, bl, out);
}
